// Attention_Luong_for_batch_56083682951388
// MI455X (gfx1250) — matrix operand order repaired and checked by reading, not compiled, not run on hardware
//
#include <hip/hip_runtime.h>
#include <hip/hip_bf16.h>
#include <math.h>

typedef __attribute__((ext_vector_type(16))) _Float16 v16h;
typedef __attribute__((ext_vector_type(8)))  float    v8f;
typedef __attribute__((ext_vector_type(4)))  float    v4f;

#define H 1024
#define B 32
#define S 2048

// ---------------------------------------------------------------------------
// Stage 1: q[b][n] = sum_o hidden[b][o] * W[o][n]   (32 x 1024 x 1024 GEMM)
// One wave (32 threads) per 16x16 output tile, K-loop in steps of 32 using
// v_wmma_f32_16x16x32_f16 (f16 inputs, f32 accumulate).
// grid = (H/16, B/16), block = 32.
// ---------------------------------------------------------------------------
__global__ void qproj_wmma(const float* __restrict__ hidden,
                           const float* __restrict__ W,
                           float* __restrict__ q) {
    const int n0   = blockIdx.x * 16;
    const int b0   = blockIdx.y * 16;
    const int lane = threadIdx.x;      // 0..31 (wave32)
    const int half = lane >> 4;        // 0: lanes 0-15, 1: lanes 16-31
    const int lid  = lane & 15;

    v8f acc = {};
    for (int k0 = 0; k0 < H; k0 += 32) {
        // A fragment: hidden rows (M = b0+lid), contiguous k order (same as B):
        //   half==0 -> K = 0..15 ; half==1 -> K = 16..31
        const float* arow = hidden + (b0 + lid) * H + k0;
        const int koff = half * 16;
        v16h a;
#pragma unroll
        for (int i = 0; i < 8; ++i) {
            a[i]     = (_Float16)arow[koff + i];
            a[8 + i] = (_Float16)arow[koff + 8 + i];
        }
        // B fragment: W[k0+kk][n0+lid], contiguous k order (same as A):
        //   half==0 -> K = 0..15 ; half==1 -> K = 16..31 ; N = lid
        const float* bcol = W + (size_t)(k0 + half * 16) * H + (n0 + lid);
        v16h bfrag;
#pragma unroll
        for (int i = 0; i < 16; ++i)
            bfrag[i] = (_Float16)bcol[(size_t)i * H];

        acc = __builtin_amdgcn_wmma_f32_16x16x32_f16(
            /*neg_a=*/false, a, /*neg_b=*/false, bfrag,
            /*c_mod=*/(short)0, acc, /*reuse_a=*/false, /*reuse_b=*/false);
    }
    // C/D layout: acc[v] -> row (v + 8*half), col lid
#pragma unroll
    for (int v = 0; v < 8; ++v) {
        const int m = v + half * 8;
        q[(b0 + m) * H + (n0 + lid)] = acc[v];
    }
}

// ---------------------------------------------------------------------------
// c[b] = hidden[b,:] . bias   (tiny: 32 dots of length 1024, one wave each)
// ---------------------------------------------------------------------------
__global__ void bias_dot(const float* __restrict__ hidden,
                         const float* __restrict__ bias,
                         float* __restrict__ c) {
    const int b = blockIdx.x;
    float s = 0.f;
    for (int h = threadIdx.x; h < H; h += 32)
        s += hidden[b * H + h] * bias[h];
#pragma unroll
    for (int off = 16; off > 0; off >>= 1)
        s += __shfl_xor(s, off, 32);
    if (threadIdx.x == 0) c[b] = s;
}

// ---------------------------------------------------------------------------
// Stage 2: scores[b][s] = enc[s,b,:] . q[b,:] + c[b]
// The 256 MB single-pass stream -> HBM-bound (~11 us at 23.3 TB/s).
// One wave per (s,b) pair; each lane does 8 non-temporal b128 loads of enc.
// q rows (128 KB total) stay cache-resident. block = 256 (8 waves).
// ---------------------------------------------------------------------------
__global__ void scores_kernel(const float* __restrict__ enc,
                              const float* __restrict__ q,
                              const float* __restrict__ c,
                              float* __restrict__ scores) {
    const int wave = threadIdx.x >> 5;
    const int lane = threadIdx.x & 31;
    const int idx  = blockIdx.x * 8 + wave;   // 0 .. S*B-1
    const int s    = idx >> 5;                // idx / B
    const int b    = idx & 31;                // idx % B

    const v4f* e  = (const v4f*)(enc + ((size_t)s * B + b) * H);
    const v4f* qr = (const v4f*)(q + (size_t)b * H);

    float acc = 0.f;
#pragma unroll
    for (int i = 0; i < 8; ++i) {
        const v4f ev = __builtin_nontemporal_load(&e[lane + i * 32]); // stream, don't cache
        const v4f qv = qr[lane + i * 32];
        acc += ev.x * qv.x + ev.y * qv.y + ev.z * qv.z + ev.w * qv.w;
    }
#pragma unroll
    for (int off = 16; off > 0; off >>= 1)
        acc += __shfl_xor(acc, off, 32);
    if (lane == 0) scores[(size_t)b * S + s] = acc + c[b];
}

// ---------------------------------------------------------------------------
// Stage 3: softmax over s per batch row. One 256-thread block per b.
// ---------------------------------------------------------------------------
__global__ void softmax_kernel(const float* __restrict__ scores,
                               float* __restrict__ out) {
    __shared__ float red[8];
    __shared__ float s_max, s_sum;
    const int b    = blockIdx.x;
    const int tid  = threadIdx.x;
    const int wave = tid >> 5;
    const int lane = tid & 31;
    const float* row = scores + (size_t)b * S;

    float m = -INFINITY;
    for (int s = tid; s < S; s += 256) m = fmaxf(m, row[s]);
#pragma unroll
    for (int off = 16; off > 0; off >>= 1) m = fmaxf(m, __shfl_xor(m, off, 32));
    if (lane == 0) red[wave] = m;
    __syncthreads();
    if (tid == 0) {
        float mm = red[0];
#pragma unroll
        for (int i = 1; i < 8; ++i) mm = fmaxf(mm, red[i]);
        s_max = mm;
    }
    __syncthreads();
    m = s_max;

    float sum = 0.f;
    for (int s = tid; s < S; s += 256) sum += __expf(row[s] - m);
#pragma unroll
    for (int off = 16; off > 0; off >>= 1) sum += __shfl_xor(sum, off, 32);
    if (lane == 0) red[wave] = sum;
    __syncthreads();
    if (tid == 0) {
        float t = 0.f;
#pragma unroll
        for (int i = 0; i < 8; ++i) t += red[i];
        s_sum = t;
    }
    __syncthreads();
    const float inv = 1.0f / s_sum;
    for (int s = tid; s < S; s += 256)
        out[(size_t)b * S + s] = __expf(row[s] - m) * inv;
}

// ---------------------------------------------------------------------------
extern "C" void kernel_launch(void* const* d_in, const int* in_sizes, int n_in,
                              void* d_out, int out_size, void* d_ws, size_t ws_size,
                              hipStream_t stream) {
    const float* hidden = (const float*)d_in[0];   // [1,B,H]
    const float* enc    = (const float*)d_in[1];   // [S,B,H]
    const float* attn_w = (const float*)d_in[2];   // [H,H]
    const float* attn_b = (const float*)d_in[3];   // [H]
    float* out = (float*)d_out;                    // [1,B,S]

    float* ws     = (float*)d_ws;
    float* q      = ws;                 // B*H   = 32768 floats
    float* c      = ws + B * H;         // B     = 32 floats (pad to 64)
    float* scores = ws + B * H + 64;    // B*S   = 65536 floats

    qproj_wmma   <<<dim3(H / 16, B / 16), 32,  0, stream>>>(hidden, attn_w, q);
    bias_dot     <<<B,                    32,  0, stream>>>(hidden, attn_b, c);
    scores_kernel<<<(S * B) / 8,          256, 0, stream>>>(enc, q, c, scores);
    softmax_kernel<<<B,                   256, 0, stream>>>(scores, out);
}
